// SparseBPNeuralNetwork_26963804684522
// MI455X (gfx1250) — matrix operand order repaired and checked by reading, not compiled, not run on hardware
//
#include <hip/hip_runtime.h>
#include <hip/hip_bf16.h>

typedef __attribute__((ext_vector_type(16))) _Float16 v16h;
typedef __attribute__((ext_vector_type(8)))  float    v8f;
typedef __attribute__((ext_vector_type(4)))  float    v4f;
typedef __attribute__((ext_vector_type(4)))  _Float16 v4h;

#define BM 128
#define BN 256
#define BK 32

union FragH { v16h h; unsigned u[8]; };

// ---------------- f32 -> f16 cast (4 elems / thread) ----------------
__global__ __launch_bounds__(256) void cast_f32_f16(const float* __restrict__ src,
                                                    _Float16* __restrict__ dst, long n4) {
    long i = (long)blockIdx.x * blockDim.x + threadIdx.x;
    if (i >= n4) return;
    v4f f = *(const v4f*)(src + i * 4);
    v4h h;
    h.x = (_Float16)f.x; h.y = (_Float16)f.y; h.z = (_Float16)f.z; h.w = (_Float16)f.w;
    *(v4h*)(dst + i * 4) = h;
}

// ------------- transpose + cast: src f32 [rows x cols] -> dst f16 [cols x rows] -------------
__global__ __launch_bounds__(256) void transpose_cast(const float* __restrict__ src,
                                                      int rows, int cols,
                                                      _Float16* __restrict__ dst) {
    __shared__ float tile[32][33];
    int r0 = blockIdx.y * 32, c0 = blockIdx.x * 32;
    int t = threadIdx.x;
    {
        int r = t >> 3, c4 = (t & 7) * 4;
        v4f v = *(const v4f*)(src + (size_t)(r0 + r) * cols + c0 + c4);
        tile[r][c4] = v.x; tile[r][c4 + 1] = v.y; tile[r][c4 + 2] = v.z; tile[r][c4 + 3] = v.w;
    }
    __syncthreads();
    {
        int c = t >> 3, r4 = (t & 7) * 4;
        v4h h;
        h.x = (_Float16)tile[r4][c];     h.y = (_Float16)tile[r4 + 1][c];
        h.z = (_Float16)tile[r4 + 2][c]; h.w = (_Float16)tile[r4 + 3][c];
        *(v4h*)(dst + (size_t)(c0 + c) * rows + r0 + r4) = h;
    }
}

// ---------------- cn_update: gather-product -> clip -> 2*atanh ----------------
template<bool TANHLOAD>
__global__ __launch_bounds__(256) void cn_kernel(const float* __restrict__ src, int srow,
                                                 const int* __restrict__ idx, int D,
                                                 float* __restrict__ dst, int H) {
    extern __shared__ float row[];
    int b = blockIdx.y;
    const float* s = src + (size_t)b * srow;
    for (int i = threadIdx.x; i < srow; i += blockDim.x) {
        float v = s[i];
        row[i] = TANHLOAD ? tanhf(0.5f * v) : v;
    }
    __syncthreads();
    int h = blockIdx.x * blockDim.x + threadIdx.x;
    if (h >= H) return;
    float prod = 1.0f;
    bool any = false;
    for (int d = 0; d < D; ++d) {
        float v = row[idx[h * D + d]];
        if (v != 0.0f) { prod *= v; any = true; }
    }
    float t = any ? prod : 0.0f;
    t = fminf(fmaxf(t, -0.999999f), 0.999999f);
    dst[(size_t)b * H + h] = 2.0f * atanhf(t);
}

// ---------------- async global->LDS staging (no data VGPRs, ASYNCcnt-tracked) ----------------
__device__ inline void async_wait0() {
    asm volatile("s_wait_asynccnt 0x0" ::: "memory");
}

// A tile: 128 x 32 halves; thread -> 32 contiguous bytes (2 x b128).
__device__ inline void ld_A_async(const _Float16* __restrict__ A, int lda, int m0, int k0,
                                  unsigned ldsBase, int tid) {
    int m = tid >> 1, koff = (tid & 1) * 16;
    unsigned long long g = (unsigned long long)(size_t)(A + (size_t)(m0 + m) * lda + k0 + koff);
    unsigned lds = ldsBase + (unsigned)(m * BK + koff) * 2u;
    asm volatile(
        "global_load_async_to_lds_b128 %0, %1, off\n\t"
        "global_load_async_to_lds_b128 %0, %1, off offset:16"
        :: "v"(lds), "v"(g) : "memory");
}

// B tile: 256 x 32 halves; thread -> 64 contiguous bytes (4 x b128).
__device__ inline void ld_B_async(const _Float16* __restrict__ B, int ldb, int n0, int k0,
                                  unsigned ldsBase, int tid) {
    int n = tid;
    unsigned long long g = (unsigned long long)(size_t)(B + (size_t)(n0 + n) * ldb + k0);
    unsigned lds = ldsBase + (unsigned)(n * BK) * 2u;
    asm volatile(
        "global_load_async_to_lds_b128 %0, %1, off\n\t"
        "global_load_async_to_lds_b128 %0, %1, off offset:16\n\t"
        "global_load_async_to_lds_b128 %0, %1, off offset:32\n\t"
        "global_load_async_to_lds_b128 %0, %1, off offset:48"
        :: "v"(lds), "v"(g) : "memory");
}

// ISA 7.12.2 wave32 fragment layouts; wave patch 64x64 -> 16 wmma per K-step.
__device__ inline void mma_step(const _Float16* As, const _Float16* Bs,
                                v8f acc[4][4], int lane, int waveM, int waveN) {
    FragH a[4], b[4];
    int ml  = lane & 15;
    int kbA = (lane >> 4) * 8;
    #pragma unroll
    for (int mi = 0; mi < 4; ++mi) {
        const _Float16* ap = &As[(waveM * 64 + mi * 16 + ml) * BK];
        #pragma unroll
        for (int v = 0; v < 8; ++v) {
            int k = ((v >> 2) << 4) + kbA + ((v & 3) << 1);
            a[mi].u[v] = *(const unsigned*)(ap + k);
        }
    }
    int nl  = lane & 15;
    int kbB = (lane >> 4) * 8;
    #pragma unroll
    for (int ni = 0; ni < 4; ++ni) {
        const _Float16* bp = &Bs[(waveN * 64 + ni * 16 + nl) * BK + kbB];
        #pragma unroll
        for (int v = 0; v < 8; ++v)
            b[ni].u[v] = *(const unsigned*)(bp + ((v >> 2) << 4) + ((v & 3) << 1));
    }
    #pragma unroll
    for (int mi = 0; mi < 4; ++mi)
        #pragma unroll
        for (int ni = 0; ni < 4; ++ni)
            acc[mi][ni] = __builtin_amdgcn_wmma_f32_16x16x32_f16(
                false, a[mi].h, false, b[ni].h, (short)0, acc[mi][ni], false, false);
}

// Double-buffered K-loop: async-issue next tile, WMMA on current, wait+barrier.
__device__ inline void gemm_loop(const _Float16* __restrict__ A, int lda,
                                 const _Float16* __restrict__ B, int ldb, int K,
                                 int m0, int n0, const _Float16* As, const _Float16* Bs,
                                 unsigned asBase, unsigned bsBase,
                                 v8f acc[4][4], int tid, int lane, int waveM, int waveN) {
    int nsteps = K / BK;
    ld_A_async(A, lda, m0, 0, asBase, tid);
    ld_B_async(B, ldb, n0, 0, bsBase, tid);
    async_wait0();
    __syncthreads();
    for (int s = 0; s < nsteps; ++s) {
        int cur = s & 1, nxt = cur ^ 1;
        bool more = (s + 1 < nsteps);
        if (more) {
            ld_A_async(A, lda, m0, (s + 1) * BK, asBase + (unsigned)(nxt * BM * BK * 2), tid);
            ld_B_async(B, ldb, n0, (s + 1) * BK, bsBase + (unsigned)(nxt * BN * BK * 2), tid);
        }
        mma_step(As + cur * (BM * BK), Bs + cur * (BN * BK), acc, lane, waveM, waveN);
        if (more) async_wait0();
        __syncthreads();
    }
}

// EPI: 0 = store f32; 1 = store tanh(0.5*acc); 2 = w = acc + addsrc, C0 = w, C1 = sigmoid(w)
template<int EPI>
__global__ __launch_bounds__(256) void gemm_wmma_nt(
    const _Float16* __restrict__ A1, int lda1, const _Float16* __restrict__ B1, int ldb1, int K1,
    const _Float16* __restrict__ A2, int lda2, const _Float16* __restrict__ B2, int ldb2, int K2,
    const float* __restrict__ addsrc, float* __restrict__ C0, float* __restrict__ C1, int ldc) {
    __shared__ __align__(16) _Float16 As[2 * BM * BK];   // 16 KB
    __shared__ __align__(16) _Float16 Bs[2 * BN * BK];   // 32 KB
    int tid = threadIdx.x;
    int wave = tid >> 5, lane = tid & 31;
    int waveM = wave & 1;    // 2 waves x 64 rows
    int waveN = wave >> 1;   // 4 waves x 64 cols
    int m0 = blockIdx.y * BM;
    int n0 = blockIdx.x * BN;
    unsigned asBase = (unsigned)(size_t)&As[0];          // LDS aperture: low 32 bits = LDS addr
    unsigned bsBase = (unsigned)(size_t)&Bs[0];

    v8f acc[4][4];
    #pragma unroll
    for (int mi = 0; mi < 4; ++mi)
        #pragma unroll
        for (int ni = 0; ni < 4; ++ni)
            acc[mi][ni] = (v8f){};

    gemm_loop(A1, lda1, B1, ldb1, K1, m0, n0, As, Bs, asBase, bsBase,
              acc, tid, lane, waveM, waveN);
    if (K2 > 0)
        gemm_loop(A2, lda2, B2, ldb2, K2, m0, n0, As, Bs, asBase, bsBase,
                  acc, tid, lane, waveM, waveN);

    int nl = lane & 15, hb = (lane >> 4) * 8;
    #pragma unroll
    for (int mi = 0; mi < 4; ++mi) {
        #pragma unroll
        for (int ni = 0; ni < 4; ++ni) {
            int col  = n0 + waveN * 64 + ni * 16 + nl;
            int rowb = m0 + waveM * 64 + mi * 16 + hb;
            #pragma unroll
            for (int r = 0; r < 8; ++r) {
                float v = acc[mi][ni][r];
                size_t o = (size_t)(rowb + r) * ldc + col;
                if (EPI == 0) {
                    C0[o] = v;
                } else if (EPI == 1) {
                    C0[o] = tanhf(0.5f * v);
                } else {
                    float w = v + addsrc[o];
                    C0[o] = w;
                    C1[o] = 1.0f / (1.0f + expf(-w));
                }
            }
        }
    }
}

// ---------------- host ----------------
static inline void cast_launch(const float* s, _Float16* d, long n, hipStream_t st) {
    long n4 = n / 4;
    cast_f32_f16<<<(unsigned)((n4 + 255) / 256), 256, 0, st>>>(s, d, n4);
}

extern "C" void kernel_launch(void* const* d_in, const int* in_sizes, int n_in,
                              void* d_out, int out_size, void* d_ws, size_t ws_size,
                              hipStream_t stream) {
    (void)in_sizes; (void)n_in; (void)out_size; (void)ws_size;
    const int Bsz = 2048, Nn = 1024, Hmax = 4096;

    const float* x_in  = (const float*)d_in[0];
    const float* S     = (const float*)d_in[1];
    const float* B1w   = (const float*)d_in[2];
    const float* B2w   = (const float*)d_in[3];
    const float* Wv1   = (const float*)d_in[4];
    const float* Wout1 = (const float*)d_in[5];
    const float* Wv2   = (const float*)d_in[6];
    const float* Wout2 = (const float*)d_in[7];
    const int*   i1f   = (const int*)d_in[8];
    const int*   i1m   = (const int*)d_in[9];
    const int*   i2f   = (const int*)d_in[10];
    const int*   i2m   = (const int*)d_in[11];
    float* out = (float*)d_out;

    char* ws = (char*)d_ws;
    size_t off = 0;
    auto take = [&](size_t bytes) { size_t o = off; off += (bytes + 255) & ~(size_t)255; return o; };
    float*    ws_x   = (float*)   (ws + take((size_t)Bsz * Nn   * 4));
    float*    ws_chS = (float*)   (ws + take((size_t)Bsz * Nn   * 4));
    float*    ws_t1  = (float*)   (ws + take((size_t)Bsz * Hmax * 4));
    float*    ws_t2  = (float*)   (ws + take((size_t)Bsz * Hmax * 4));
    _Float16* xh     = (_Float16*)(ws + take((size_t)Bsz * Nn   * 2));
    _Float16* chSh   = (_Float16*)(ws + take((size_t)Bsz * Nn   * 2));
    _Float16* th     = (_Float16*)(ws + take((size_t)Bsz * Hmax * 2));
    _Float16* t2h    = (_Float16*)(ws + take((size_t)Bsz * Hmax * 2));
    _Float16* ShT    = (_Float16*)(ws + take((size_t)Nn  * Nn   * 2));   // S[k]^T, N-major
    _Float16* BmT    = (_Float16*)(ws + take((size_t)Nn  * Hmax * 2));   // Bm^T,  H-major
    _Float16* Wvh    = (_Float16*)(ws + take((size_t)Hmax* Hmax * 2));   // already N-major
    _Float16* Woh    = (_Float16*)(ws + take((size_t)Nn  * Hmax * 2));   // already N-major

    for (int k = 0; k < 10; ++k) {
        bool g1 = (k % 2) == 0;
        int j = k / 2;
        int H = g1 ? 3072 : 4096;
        int D = g1 ? 6 : 8;
        const float* Wv  = g1 ? (Wv1   + (size_t)j * H * H)  : (Wv2   + (size_t)j * H * H);
        const float* Wo  = g1 ? (Wout1 + (size_t)j * Nn * H) : (Wout2 + (size_t)j * Nn * H);
        const float* Bm  = g1 ? B1w : B2w;
        const int*   idf = g1 ? i1f : i2f;
        const int*   idm = g1 ? i1m : i2m;
        const float* xsrc = (k == 0) ? x_in : ws_x;

        // Stage operands: S[k]^T and Bm^T so every GEMM B-load is the fast NT path.
        transpose_cast<<<dim3(Nn / 32, Nn / 32), 256, 0, stream>>>(
            S + (size_t)k * Nn * Nn, Nn, Nn, ShT);
        transpose_cast<<<dim3(H / 32, Nn / 32), 256, 0, stream>>>(Bm, Nn, H, BmT);
        cast_launch(xsrc, xh, (long)Bsz * Nn, stream);

        // chS = x @ S[k]
        gemm_wmma_nt<0><<<dim3(Nn / BN, Bsz / BM), 256, 0, stream>>>(
            xh, Nn, ShT, Nn, Nn,
            nullptr, 0, nullptr, 0, 0,
            nullptr, ws_chS, nullptr, Nn);

        // variable-node stage 1: tanh -> gather-product -> clip -> 2*atanh
        cn_kernel<true><<<dim3(H / 256, Bsz), 256, Nn * sizeof(float), stream>>>(
            xsrc, Nn, idf, D, ws_t1, H);

        // t = tanh(0.5*(t1 @ Wv^T + chS @ Bm)), fused dual-K GEMM
        cast_launch(ws_t1, th, (long)Bsz * H, stream);
        cast_launch(ws_chS, chSh, (long)Bsz * Nn, stream);
        cast_launch(Wv, Wvh, (long)H * H, stream);
        gemm_wmma_nt<1><<<dim3(H / BN, Bsz / BM), 256, 0, stream>>>(
            th, H, Wvh, H, H,
            chSh, Nn, BmT, Nn, Nn,
            nullptr, ws_t1, nullptr, H);   // ws_t1 now holds tanh'd pre-cn values

        // variable-node stage 2 (source already tanh'd)
        cn_kernel<false><<<dim3(H / 256, Bsz), 256, H * sizeof(float), stream>>>(
            ws_t1, H, idm, D, ws_t2, H);

        // x = t2 @ Wo^T + chS ; emit sigmoid(x) to d_out[k]
        cast_launch(ws_t2, t2h, (long)Bsz * H, stream);
        cast_launch(Wo, Woh, (long)Nn * H, stream);
        gemm_wmma_nt<2><<<dim3(Nn / BN, Bsz / BM), 256, 0, stream>>>(
            t2h, H, Woh, H, H,
            nullptr, 0, nullptr, 0, 0,
            ws_chS, ws_x, out + (size_t)k * Bsz * Nn, Nn);
    }
}
